// Model_Net_62088047231034
// MI455X (gfx1250) — matrix operand order repaired and checked by reading, not compiled, not run on hardware
//
#include <hip/hip_runtime.h>
#include <hip/hip_bf16.h>
#include <stdint.h>

// ---------------- problem constants ----------------
static constexpr int NN   = 40000;   // nodes
static constexpr int EE   = 640000;  // edges
static constexpr int FIN  = 1280;    // input features
static constexpr int HH   = 128;     // hidden dim
static constexpr int GG   = 64;      // graphs
static constexpr int OUTD = 273;     // output dim
#define NEG_SLOPE 0.2f
#define BN_EPS    1e-5f

// ---------------- types ----------------
typedef __attribute__((ext_vector_type(16))) __bf16 v16bf;
typedef __attribute__((ext_vector_type(8)))  float  v8f;

__device__ __forceinline__ unsigned int f32_to_bf16_bits(float f) {
    unsigned int u = __float_as_uint(f);
    unsigned int r = u + 0x7FFFu + ((u >> 16) & 1u);   // round-to-nearest-even
    return r >> 16;
}
__device__ __forceinline__ unsigned int pack_bf16(float lo, float hi) {
    return f32_to_bf16_bits(lo) | (f32_to_bf16_bits(hi) << 16);
}

// monotone order-preserving float<->uint key (for atomic max on signed floats)
__device__ __forceinline__ unsigned int fkey(float f) {
    unsigned int u = __float_as_uint(f);
    return (u & 0x80000000u) ? ~u : (u | 0x80000000u);
}
__device__ __forceinline__ float funkey(unsigned int k) {
    unsigned int u = (k & 0x80000000u) ? (k & 0x7FFFFFFFu) : ~k;
    return __uint_as_float(u);
}
#define NEG_INF_KEY 0x007FFFFFu   // fkey(-inf)

// ============================================================
// WMMA bf16 GEMM:  C[M x 128] = A[M x K] * B[K x 128]
// A,B are f32 in global; converted to bf16 while staging to LDS.
// Block = 256 threads (8 waves). Block tile: 64 rows x 128 cols.
// Wave w: M-tile (w>>1), N-tiles 4*(w&1) .. +3.  K step = 32.
// ============================================================
__global__ __launch_bounds__(256) void gemm_wmma_bf16(
    const float* __restrict__ A, const float* __restrict__ B,
    float* __restrict__ C, int K)
{
    __shared__ unsigned int ldsA[64 * 16];    // [row][kpair] : 64 x 32 bf16
    __shared__ unsigned int ldsB[128 * 16];   // [n][kpair]   : pre-packed K-pairs

    const int t     = threadIdx.x;
    const int lane  = t & 31;
    const int wave  = t >> 5;
    const int m0    = blockIdx.x * 64;
    const int mtile = wave >> 1;          // 0..3
    const int ngrp  = (wave & 1) * 4;     // base N-tile (0 or 4)

    v8f acc[4] = {};                      // 4 N-tiles of 16x16 f32

    const int khalf = lane >> 4;          // A/B lane halves
    const int mloc  = lane & 15;
    const int nloc  = lane & 15;

    for (int k0 = 0; k0 < K; k0 += 32) {
        __syncthreads();
        // ---- stage A tile: 64 rows x 32 cols (f32 -> packed bf16) ----
        {
            const int row = t >> 2;
            const int cq  = t & 3;                       // 8 floats each
            const float* ap = A + (size_t)(m0 + row) * K + k0 + cq * 8;
            float4 f0 = *(const float4*)ap;
            float4 f1 = *(const float4*)(ap + 4);
            unsigned int* d = &ldsA[row * 16 + cq * 4];
            d[0] = pack_bf16(f0.x, f0.y);
            d[1] = pack_bf16(f0.z, f0.w);
            d[2] = pack_bf16(f1.x, f1.y);
            d[3] = pack_bf16(f1.z, f1.w);
        }
        // ---- stage B tile: 32 rows x 128 cols, packed as (n, kpair) ----
        {
            const int kp = t & 15;
            const int n0 = (t >> 4) * 8;
            const float* b0 = B + (size_t)(k0 + 2 * kp) * HH + n0;
            const float* b1 = b0 + HH;
            float4 x0 = *(const float4*)b0, x1 = *(const float4*)(b0 + 4);
            float4 y0 = *(const float4*)b1, y1 = *(const float4*)(b1 + 4);
            float lo[8] = {x0.x, x0.y, x0.z, x0.w, x1.x, x1.y, x1.z, x1.w};
            float hi[8] = {y0.x, y0.y, y0.z, y0.w, y1.x, y1.y, y1.z, y1.w};
#pragma unroll
            for (int i = 0; i < 8; i++)
                ldsB[(n0 + i) * 16 + kp] = pack_bf16(lo[i], hi[i]);
        }
        __syncthreads();

        // ---- A fragment (16x32 bf16, ISA layout) ----
        union { unsigned int u[8]; v16bf v; } af;
#pragma unroll
        for (int r = 0; r < 8; r++) {
            const int g = r >> 2, w = r & 3;
            af.u[r] = ldsA[(mtile * 16 + mloc) * 16 + g * 8 + khalf * 4 + w];
        }
        // ---- 4 x (B fragment + WMMA) ----
#pragma unroll
        for (int j = 0; j < 4; j++) {
            union { unsigned int u[8]; v16bf v; } bf;
            const int ncol = (ngrp + j) * 16 + nloc;
#pragma unroll
            for (int r = 0; r < 8; r++)
                bf.u[r] = ldsB[ncol * 16 + (r >> 2) * 8 + khalf * 4 + (r & 3)];
            acc[j] = __builtin_amdgcn_wmma_f32_16x16x32_bf16(
                false, af.v, false, bf.v, (short)0, acc[j], false, false);
        }
    }

    // ---- store C (16x16 f32 layout: VGPR r -> M = mhalf*8 + r) ----
    const int mhalf = lane >> 4;
#pragma unroll
    for (int j = 0; j < 4; j++) {
        const int col = (ngrp + j) * 16 + nloc;
#pragma unroll
        for (int r = 0; r < 8; r++) {
            const int row = m0 + mtile * 16 + mhalf * 8 + r;
            C[(size_t)row * HH + col] = acc[j][r];
        }
    }
}

// ============================================================
// helper kernels
// ============================================================
__global__ void fill_u32(unsigned int* p, unsigned int v, int n) {
    int i = blockIdx.x * blockDim.x + threadIdx.x;
    if (i < n) p[i] = v;
}

// s_src[n] = h[n]·a_src ; s_dst[n] = h[n]·a_dst
__global__ void node_dots(const float* __restrict__ h,
                          const float* __restrict__ a_s, const float* __restrict__ a_d,
                          float* __restrict__ ssrc, float* __restrict__ sdst, int n) {
    int i = blockIdx.x * blockDim.x + threadIdx.x;
    if (i >= n) return;
    const float* hp = h + (size_t)i * HH;
    float s1 = 0.f, s2 = 0.f;
#pragma unroll 4
    for (int k = 0; k < HH; k++) { float v = hp[k]; s1 += v * a_s[k]; s2 += v * a_d[k]; }
    ssrc[i] = s1; sdst[i] = s2;
}

// e = lrelu(s_src[src]+s_dst[dst]); segment max into mkey[dst]
__global__ void edge_max(const int* __restrict__ src, const int* __restrict__ dst,
                         const float* __restrict__ ssrc, const float* __restrict__ sdst,
                         float* __restrict__ ebuf, unsigned int* __restrict__ mkey, int ne) {
    int i = blockIdx.x * blockDim.x + threadIdx.x;
    if (i >= ne) return;
    float e = ssrc[src[i]] + sdst[dst[i]];
    e = (e > 0.f) ? e : e * NEG_SLOPE;
    ebuf[i] = e;
    atomicMax(&mkey[dst[i]], fkey(e));
}

// p = exp(e - m[dst]); denom[dst] += p   (p overwrites e in-place)
__global__ void edge_expsum(const int* __restrict__ dst, float* __restrict__ ebuf,
                            const unsigned int* __restrict__ mkey,
                            float* __restrict__ denom, int ne) {
    int i = blockIdx.x * blockDim.x + threadIdx.x;
    if (i >= ne) return;
    int d = dst[i];
    float p = __expf(ebuf[i] - funkey(mkey[d]));
    ebuf[i] = p;
    atomicAdd(&denom[d], p);
}

// out[dst] += alpha * h[src]  — one wave per edge, 4 floats per lane
__global__ void edge_scatter(const int* __restrict__ src, const int* __restrict__ dst,
                             const float* __restrict__ ebuf, const float* __restrict__ denom,
                             const float* __restrict__ h, float* __restrict__ outp, int ne) {
    int w = (blockIdx.x * blockDim.x + threadIdx.x) >> 5;
    int lane = threadIdx.x & 31;
    if (w >= ne) return;
    int s = src[w], d = dst[w];
    float alpha = ebuf[w] / (denom[d] + 1e-16f);
    const float4 hv = *(const float4*)(h + (size_t)s * HH + lane * 4);
    float* o = outp + (size_t)d * HH + lane * 4;
    atomicAdd(o + 0, alpha * hv.x);
    atomicAdd(o + 1, alpha * hv.y);
    atomicAdd(o + 2, alpha * hv.z);
    atomicAdd(o + 3, alpha * hv.w);
}

__global__ void bias_act(float* __restrict__ h, const float* __restrict__ b,
                         int total, int relu) {
    int i = blockIdx.x * blockDim.x + threadIdx.x;
    if (i >= total) return;
    float v = h[i] + b[i & (HH - 1)];
    if (relu) v = fmaxf(v, 0.f);
    h[i] = v;
}

__global__ void count_nodes(const int* __restrict__ batch, float* __restrict__ cnt, int n) {
    int i = blockIdx.x * blockDim.x + threadIdx.x;
    if (i < n) atomicAdd(&cnt[batch[i]], 1.0f);
}

__global__ void pool_accum(const float* __restrict__ h, const int* __restrict__ batch,
                           float* __restrict__ pooled, int total) {
    int i = blockIdx.x * blockDim.x + threadIdx.x;
    if (i >= total) return;
    int node = i >> 7, c = i & (HH - 1);
    atomicAdd(&pooled[batch[node] * HH + c], h[i]);
}

__global__ void pool_div(float* __restrict__ pooled, const float* __restrict__ cnt) {
    int i = blockIdx.x * blockDim.x + threadIdx.x;
    if (i >= GG * HH) return;
    pooled[i] /= fmaxf(cnt[i >> 7], 1.0f);
}

// z = relu(BN(pooled @ fc1_W + fc1_b))
__global__ void fc1_bn_relu(const float* __restrict__ pooled, const float* __restrict__ W,
                            const float* __restrict__ b, const float* __restrict__ g,
                            const float* __restrict__ bb, const float* __restrict__ rm,
                            const float* __restrict__ rv, float* __restrict__ z) {
    int i = blockIdx.x * blockDim.x + threadIdx.x;
    if (i >= GG * 1024) return;
    int gr = i >> 10, j = i & 1023;
    const float* p = pooled + gr * HH;
    float s = b[j];
#pragma unroll 4
    for (int k = 0; k < HH; k++) s += p[k] * W[k * 1024 + j];
    s = (s - rm[j]) * rsqrtf(rv[j] + BN_EPS) * g[j] + bb[j];
    z[i] = fmaxf(s, 0.f);
}

// out = sigmoid(z @ fc2_W + fc2_b)
__global__ void fc2_sigmoid(const float* __restrict__ z, const float* __restrict__ W,
                            const float* __restrict__ b, float* __restrict__ outp) {
    int i = blockIdx.x * blockDim.x + threadIdx.x;
    if (i >= GG * OUTD) return;
    int gr = i / OUTD, j = i - gr * OUTD;
    const float* zp = z + gr * 1024;
    float s = b[j];
#pragma unroll 4
    for (int k = 0; k < 1024; k++) s += zp[k] * W[k * OUTD + j];
    outp[i] = 1.0f / (1.0f + __expf(-s));
}

// ============================================================
// launch
// ============================================================
extern "C" void kernel_launch(void* const* d_in, const int* in_sizes, int n_in,
                              void* d_out, int out_size, void* d_ws, size_t ws_size,
                              hipStream_t stream) {
    const float* x        = (const float*)d_in[0];
    const int*   eidx     = (const int*)d_in[1];
    const int*   batch    = (const int*)d_in[2];
    const float* W1       = (const float*)d_in[3];
    const float* b1       = (const float*)d_in[4];
    const float* as1      = (const float*)d_in[5];
    const float* ad1      = (const float*)d_in[6];
    const float* W2       = (const float*)d_in[7];
    const float* b2       = (const float*)d_in[8];
    const float* as2      = (const float*)d_in[9];
    const float* ad2      = (const float*)d_in[10];
    const float* W3       = (const float*)d_in[11];
    const float* b3       = (const float*)d_in[12];
    const float* as3      = (const float*)d_in[13];
    const float* ad3      = (const float*)d_in[14];
    const float* fc1_W    = (const float*)d_in[15];
    const float* fc1_b    = (const float*)d_in[16];
    const float* bn_g     = (const float*)d_in[17];
    const float* bn_b     = (const float*)d_in[18];
    const float* bn_rm    = (const float*)d_in[19];
    const float* bn_rv    = (const float*)d_in[20];
    const float* fc2_W    = (const float*)d_in[21];
    const float* fc2_b    = (const float*)d_in[22];
    float*       out      = (float*)d_out;

    const int* src = eidx;          // edge_index[0]
    const int* dst = eidx + EE;     // edge_index[1]

    // workspace layout (floats)
    float* ws     = (float*)d_ws;
    float* hW     = ws;                         // NN*HH : pre-aggregation h = in @ W
    float* hA     = hW   + (size_t)NN * HH;     // NN*HH : aggregated output
    float* ssrc   = hA   + (size_t)NN * HH;     // NN
    float* sdst   = ssrc + NN;                  // NN
    unsigned int* mkey = (unsigned int*)(sdst + NN);   // NN (max keys)
    float* denom  = (float*)mkey + NN;          // NN
    float* ebuf   = denom + NN;                 // EE
    float* cnt    = ebuf + EE;                  // GG
    float* pooled = cnt + GG;                   // GG*HH
    float* zbuf   = pooled + (size_t)GG * HH;   // GG*1024

    const int TB = 256;
    auto run_layer = [&](const float* in, int K, const float* W, const float* b,
                         const float* a_s, const float* a_d, int relu) {
        gemm_wmma_bf16<<<NN / 64, 256, 0, stream>>>(in, W, hW, K);
        node_dots<<<(NN + TB - 1) / TB, TB, 0, stream>>>(hW, a_s, a_d, ssrc, sdst, NN);
        fill_u32<<<(NN + TB - 1) / TB, TB, 0, stream>>>(mkey, NEG_INF_KEY, NN);
        fill_u32<<<(NN + TB - 1) / TB, TB, 0, stream>>>((unsigned int*)denom, 0u, NN);
        fill_u32<<<(NN * HH) / TB, TB, 0, stream>>>((unsigned int*)hA, 0u, NN * HH);
        edge_max<<<(EE + TB - 1) / TB, TB, 0, stream>>>(src, dst, ssrc, sdst, ebuf, mkey, EE);
        edge_expsum<<<(EE + TB - 1) / TB, TB, 0, stream>>>(dst, ebuf, mkey, denom, EE);
        edge_scatter<<<EE / 8, TB, 0, stream>>>(src, dst, ebuf, denom, hW, hA, EE);
        bias_act<<<(NN * HH) / TB, TB, 0, stream>>>(hA, b, NN * HH, relu);
    };

    run_layer(x,  FIN, W1, b1, as1, ad1, 1);
    run_layer(hA, HH,  W2, b2, as2, ad2, 1);
    run_layer(hA, HH,  W3, b3, as3, ad3, 0);

    // global mean pool per graph
    fill_u32<<<(GG * (HH + 1) + TB - 1) / TB, TB, 0, stream>>>((unsigned int*)cnt, 0u, GG * (HH + 1));
    count_nodes<<<(NN + TB - 1) / TB, TB, 0, stream>>>(batch, cnt, NN);
    pool_accum<<<(NN * HH) / TB, TB, 0, stream>>>(hA, batch, pooled, NN * HH);
    pool_div<<<(GG * HH) / TB, TB, 0, stream>>>(pooled, cnt);

    // head
    fc1_bn_relu<<<(GG * 1024) / TB, TB, 0, stream>>>(pooled, fc1_W, fc1_b,
                                                     bn_g, bn_b, bn_rm, bn_rv, zbuf);
    fc2_sigmoid<<<(GG * OUTD + TB - 1) / TB, TB, 0, stream>>>(zbuf, fc2_W, fc2_b, out);
}
